// MultiHeadAttentionModule_42588895707483
// MI455X (gfx1250) — compile-verified, matrix operand order checked by reading, not run on hardware
//
#include <hip/hip_runtime.h>
#include <hip/hip_bf16.h>

// ---------------------------------------------------------------------------
// MHA for MI455X (gfx1250): H=8 heads, D=256, B=2, S=4096.
// All matmuls via v_wmma_f32_16x16x32_f16 (wave32).
// Pipeline: qkv projection -> flash attention -> output projection.
// Flash loop is double-buffered: K tile DMA'd with
// GLOBAL_LOAD_ASYNC_TO_LDS_B128 (ASYNCcnt) for block i+1 while block i's
// WMMAs run; V tile staged transposed (padded rows, conflict-free b128 reads)
// via registers so its global loads also overlap compute.
// ---------------------------------------------------------------------------

typedef __attribute__((ext_vector_type(16))) _Float16 v16h;
typedef __attribute__((ext_vector_type(8)))  float    v8f;
typedef _Float16 half_t;

constexpr int H_ = 8;
constexpr int D_ = 256;
constexpr int B_ = 2;
constexpr int S_ = 4096;
constexpr int VLD_ = 40;           // padded row length (halves) of transposed V
constexpr float SCALE_ = 0.0625f;  // 1/sqrt(256)

__device__ inline v8f wmma_f16(v16h a, v16h b, v8f c) {
    // 8 args: (neg_a, A, neg_b, B, c_mod, C, reuse_a, reuse_b)
    return __builtin_amdgcn_wmma_f32_16x16x32_f16(
        false, a, false, b, (short)0, c, false, false);
}

// --- Fragment loaders -------------------------------------------------------
// ISA 16-bit A/B layout (wave32): lane&15 selects row (A) / column (B);
// lane>>4 selects K-halves: half0 = K {0..7},{16..23}; half1 = {8..15},{24..31}.
// element (row,k) = src[row*ld + k]  (contiguous-over-k case -> b128 loads).

__device__ inline v16h frag_load_f32(const float* __restrict__ src, int ld,
                                     int row0, int k0) {
    const int lane = threadIdx.x & 31;
    const float* p = src + (size_t)(row0 + (lane & 15)) * ld
                         + k0 + ((lane >> 4) << 3);
    v16h f;
#pragma unroll
    for (int i = 0; i < 8; ++i) f[i] = (half_t)p[i];
#pragma unroll
    for (int i = 0; i < 8; ++i) f[8 + i] = (half_t)p[16 + i];
    return f;
}

__device__ inline v16h frag_load_f16(const half_t* __restrict__ src, int ld,
                                     int row0, int k0) {
    const int lane = threadIdx.x & 31;
    const half_t* p = src + (size_t)(row0 + (lane & 15)) * ld
                          + k0 + ((lane >> 4) << 3);
    v16h f;
#pragma unroll
    for (int i = 0; i < 8; ++i) f[i] = p[i];
#pragma unroll
    for (int i = 0; i < 8; ++i) f[8 + i] = p[16 + i];
    return f;
}

// ---------------------------------------------------------------------------
// Kernel 1: per-head linear projection.
//   out[h,b,s,e] = sum_d X[b,s,d] * W[h,e,d] + bias[h,e]   (stored as f16)
// Grid: (S/64, D/64, H*B). Block: 256 (8 waves); wave -> 16x32 subtile.
// ---------------------------------------------------------------------------
__global__ __launch_bounds__(256) void mha_proj_kernel(
    const float* __restrict__ X,    // [B,S,D]
    const float* __restrict__ W,    // [H,D,D]
    const float* __restrict__ bias, // [H,D]
    half_t* __restrict__ out)       // [H,B,S,D]
{
    const int wave = threadIdx.x >> 5;
    const int lane = threadIdx.x & 31;
    const int h = blockIdx.z / B_;
    const int b = blockIdx.z % B_;
    const int m0 = blockIdx.x * 64 + (wave >> 1) * 16; // s-rows
    const int n0 = blockIdx.y * 64 + (wave & 1) * 32;  // e-cols

    const float* x = X + (size_t)b * S_ * D_;
    const float* w = W + (size_t)h * D_ * D_;

    v8f c0 = {}, c1 = {};
#pragma unroll
    for (int k = 0; k < D_; k += 32) {
        v16h a  = frag_load_f32(x, D_, m0, k);
        v16h b0 = frag_load_f32(w, D_, n0, k);
        v16h b1 = frag_load_f32(w, D_, n0 + 16, k);
        c0 = wmma_f16(a, b0, c0);
        c1 = wmma_f16(a, b1, c1);
    }

    const float* bh = bias + (size_t)h * D_;
    half_t* o = out + ((size_t)h * B_ + b) * S_ * D_;
    const int colA = n0 + (lane & 15);
    const int colB = colA + 16;
    const int rb   = m0 + ((lane >> 4) << 3);
    const float biasA = bh[colA];
    const float biasB = bh[colB];
#pragma unroll
    for (int r = 0; r < 8; ++r) {
        o[(size_t)(rb + r) * D_ + colA] = (half_t)(c0[r] + biasA);
        o[(size_t)(rb + r) * D_ + colB] = (half_t)(c1[r] + biasB);
    }
}

// --- attention staging helpers ---------------------------------------------

// Issue async DMA of one 32x256 f16 K tile into LDS (16B per lane-op).
__device__ inline void stage_k_async(const half_t* gk, unsigned ldsByteOff) {
    const uintptr_t kg = (uintptr_t)gk;
    for (int i = threadIdx.x; i < (32 * D_) / 8; i += 128) {
        const unsigned ldst = ldsByteOff + (unsigned)i * 16u;
        const unsigned long long gsrc =
            (unsigned long long)(kg + (size_t)i * 16u);
        asm volatile("global_load_async_to_lds_b128 %0, %1, off"
                     :: "v"(ldst), "v"(gsrc) : "memory");
    }
}

// Load one 32x256 f16 V tile into registers (8 x uint4 per thread).
__device__ inline void load_v_regs(const half_t* gv, uint4* vreg) {
    const uint4* vs = (const uint4*)gv;
#pragma unroll
    for (int j = 0; j < 8; ++j) vreg[j] = vs[threadIdx.x + j * 128];
}

// Scatter V registers transposed into lVt[e][t] (row pad VLD_ halves).
__device__ inline void scatter_v(const uint4* vreg, half_t* lVt) {
#pragma unroll
    for (int j = 0; j < 8; ++j) {
        const int i  = threadIdx.x + j * 128;
        const int t  = i >> 5;          // 32 uint4 per 256-wide row
        const int e0 = (i & 31) << 3;
        const half_t* hh = (const half_t*)&vreg[j];
#pragma unroll
        for (int jj = 0; jj < 8; ++jj)
            lVt[(e0 + jj) * VLD_ + t] = hh[jj];
    }
}

// ---------------------------------------------------------------------------
// Kernel 2: flash attention over one (h,b), 64 query rows per workgroup.
// Block: 128 threads = 4 waves; wave owns 16 query rows.
// Double-buffered 32-row K/V blocks; staging of block i+1 (async K DMA +
// V loads-to-registers) overlaps the WMMA work on block i. One barrier/iter.
// ---------------------------------------------------------------------------
__global__ __launch_bounds__(128) void mha_attn_kernel(
    const half_t* __restrict__ Q,  // [H,B,S,D]
    const half_t* __restrict__ K,
    const half_t* __restrict__ V,
    half_t* __restrict__ O)        // [H,B,S,D]
{
    __shared__ half_t lK[2][32 * D_];     // 2 x 16 KB, row-major [t][d]
    __shared__ half_t lVt[2][D_ * VLD_];  // 2 x 20 KB, transposed [e][t]
    __shared__ half_t lP[4][16 * 32];     //      4 KB, per-wave P patch

    const int wave = threadIdx.x >> 5;
    const int lane = threadIdx.x & 31;
    const int qrow0 = blockIdx.x * 64 + wave * 16;  // global query row
    const size_t base = (size_t)blockIdx.y * S_ * D_;

    const half_t* q  = Q + base;
    const half_t* kp = K + base;
    const half_t* vp = V + base;

    // Generic->LDS aperture: low 32 bits of a __shared__ pointer are the LDS
    // byte offset (ISA 10.2).
    const unsigned lK0off = (unsigned)(uintptr_t)(&lK[0][0]);
    const unsigned lKstride = 32u * D_ * 2u;   // bytes per K buffer

    // Q fragments for this wave's 16 rows, kept in registers (8 x v16h).
    v16h qa[8];
#pragma unroll
    for (int kk = 0; kk < 8; ++kk)
        qa[kk] = frag_load_f16(q, D_, qrow0, kk * 32);

    float row_m[8], row_l[8];
    v8f acc[16];
#pragma unroll
    for (int r = 0; r < 8; ++r) { row_m[r] = -3.0e38f; row_l[r] = 0.0f; }
#pragma unroll
    for (int n = 0; n < 16; ++n) acc[n] = v8f{0,0,0,0,0,0,0,0};

    // ---- prologue: stage block 0 into buffer 0 ----
    {
        uint4 vreg[8];
        stage_k_async(kp, lK0off);
        load_v_regs(vp, vreg);
        scatter_v(vreg, &lVt[0][0]);
    }
    asm volatile("s_wait_asynccnt 0x0" ::: "memory");
    __syncthreads();

    constexpr int NB = S_ / 32;   // 128 key blocks
    for (int ib = 0; ib < NB; ++ib) {
        const int cur = ib & 1;
        const int nxt = cur ^ 1;
        const bool more = (ib + 1) < NB;   // uniform branch: EXEC stays full

        // ---- stage block ib+1 (in flight during this block's WMMAs) ----
        uint4 vreg[8];
        if (more) {
            const size_t kt = (size_t)(ib + 1) * 32;
            stage_k_async(kp + kt * D_, lK0off + (unsigned)nxt * lKstride);
            load_v_regs(vp + kt * D_, vreg);
            if (ib + 2 < NB) {  // prefetch block ib+2 toward L2
                __builtin_prefetch(kp + (kt + 32) * D_ + lane * 64, 0, 1);
                __builtin_prefetch(vp + (kt + 32) * D_ + lane * 64, 0, 1);
            }
        }

        const half_t* lkc = &lK[cur][0];
        const half_t* lvc = &lVt[cur][0];

        // Scores: S = Q(16x256) x K_blk^T -> 16x32, 2 N-frags, 8 K-steps.
        v8f s0 = {}, s1 = {};
#pragma unroll
        for (int kk = 0; kk < 8; ++kk) {
            s0 = wmma_f16(qa[kk], frag_load_f16(lkc, D_, 0,  kk * 32), s0);
            s1 = wmma_f16(qa[kk], frag_load_f16(lkc, D_, 16, kk * 32), s1);
        }

        // Online softmax. C layout: VGPR r <-> row r + 8*(lane>=16);
        // row reduction = xor-shuffles within each 16-lane half.
        float mnew[8];
#pragma unroll
        for (int r = 0; r < 8; ++r) {
            s0[r] *= SCALE_;
            s1[r] *= SCALE_;
            float mx = fmaxf(row_m[r], fmaxf(s0[r], s1[r]));
#pragma unroll
            for (int off = 1; off < 16; off <<= 1)
                mx = fmaxf(mx, __shfl_xor(mx, off, 32));
            mnew[r] = mx;
        }
#pragma unroll
        for (int r = 0; r < 8; ++r) {
            const float corr = __expf(row_m[r] - mnew[r]);
            const float p0 = __expf(s0[r] - mnew[r]);
            const float p1 = __expf(s1[r] - mnew[r]);
            s0[r] = p0;
            s1[r] = p1;
            float rs = p0 + p1;
#pragma unroll
            for (int off = 1; off < 16; off <<= 1)
                rs += __shfl_xor(rs, off, 32);
            row_l[r] = row_l[r] * corr + rs;
            row_m[r] = mnew[r];
#pragma unroll
            for (int n = 0; n < 16; ++n) acc[n][r] *= corr;
        }

        // P (C-layout) -> per-wave LDS patch -> A-layout fragment.
        // Same-wave LDS ops are in-order: wave-local s_wait_dscnt suffices.
        half_t* pw = &lP[wave][0];
        {
            const int colb = lane & 15;
            const int rowb = (lane >> 4) << 3;
#pragma unroll
            for (int r = 0; r < 8; ++r) {
                pw[(rowb + r) * 32 + colb]      = (half_t)s0[r];
                pw[(rowb + r) * 32 + 16 + colb] = (half_t)s1[r];
            }
        }
        asm volatile("s_wait_dscnt 0x0" ::: "memory");

        // O += P(16x32) x V_blk(32x256): 16 N-frags, single K-step.
        // B(k=t, n=e) = V[t,e] = lVt[e*VLD_ + t]  -> contiguous b128 reads,
        // 20-dword row stride => 16 lanes hit 16 distinct banks.
        v16h pa = frag_load_f16(pw, 32, 0, 0);
#pragma unroll
        for (int n = 0; n < 16; ++n)
            acc[n] = wmma_f16(pa, frag_load_f16(lvc, VLD_, n * 16, 0), acc[n]);

        // ---- finish staging block ib+1, publish, recycle buffers ----
        if (more) scatter_v(vreg, &lVt[nxt][0]);
        asm volatile("s_wait_asynccnt 0x0" ::: "memory");
        __syncthreads();
    }

    // Epilogue: O /= row_l, store f16.
    half_t* o = O + base;
    const int colb = lane & 15;
    const int rb   = qrow0 + ((lane >> 4) << 3);
#pragma unroll
    for (int r = 0; r < 8; ++r) {
        const float inv = 1.0f / row_l[r];
#pragma unroll
        for (int n = 0; n < 16; ++n)
            o[(size_t)(rb + r) * D_ + n * 16 + colb] = (half_t)(acc[n][r] * inv);
    }
}

// ---------------------------------------------------------------------------
// Kernel 3: output projection.
//   Out[b,s,e] = sum_{h,d} attn[h,b,s,d] * Wo[e, h*D+d] + bo[e]
// Grid: (B*S/64, D/64). Block: 256 (8 waves); wave -> 16x32 subtile.
// ---------------------------------------------------------------------------
__global__ __launch_bounds__(256) void mha_outproj_kernel(
    const half_t* __restrict__ A,   // [H,B,S,D] attention outputs (f16)
    const float*  __restrict__ Wo,  // [D, H*D]
    const float*  __restrict__ bo,  // [D]
    float* __restrict__ Out)        // [B,S,D]
{
    const int wave = threadIdx.x >> 5;
    const int lane = threadIdx.x & 31;
    const int m0 = blockIdx.x * 64 + (wave >> 1) * 16; // row over B*S
    const int n0 = blockIdx.y * 64 + (wave & 1) * 32;  // e-cols
    const int b  = m0 / S_;   // 64-row tile never crosses batch boundary
    const int s0 = m0 % S_;

    v8f c0 = {}, c1 = {};
#pragma unroll 4
    for (int kc = 0; kc < H_ * D_; kc += 32) {
        const int h  = kc >> 8;      // kc / D_
        const int d0 = kc & (D_ - 1);
        const half_t* ah = A + ((size_t)h * B_ + b) * S_ * D_;
        v16h af = frag_load_f16(ah, D_, s0, d0);
        v16h b0 = frag_load_f32(Wo, H_ * D_, n0, kc);
        v16h b1 = frag_load_f32(Wo, H_ * D_, n0 + 16, kc);
        c0 = wmma_f16(af, b0, c0);
        c1 = wmma_f16(af, b1, c1);
    }

    const int colA = n0 + (lane & 15);
    const int colB = colA + 16;
    const int rb   = m0 + ((lane >> 4) << 3);
    const float biasA = bo[colA];
    const float biasB = bo[colB];
#pragma unroll
    for (int r = 0; r < 8; ++r) {
        Out[(size_t)(rb + r) * D_ + colA] = c0[r] + biasA;
        Out[(size_t)(rb + r) * D_ + colB] = c1[r] + biasB;
    }
}

// ---------------------------------------------------------------------------
extern "C" void kernel_launch(void* const* d_in, const int* in_sizes, int n_in,
                              void* d_out, int out_size, void* d_ws, size_t ws_size,
                              hipStream_t stream) {
    (void)in_sizes; (void)n_in; (void)out_size; (void)ws_size;

    const float* query = (const float*)d_in[0];
    const float* key_  = (const float*)d_in[1];
    const float* value = (const float*)d_in[2];
    const float* Wq    = (const float*)d_in[3];
    const float* bq    = (const float*)d_in[4];
    const float* Wk    = (const float*)d_in[5];
    const float* bk    = (const float*)d_in[6];
    const float* Wv    = (const float*)d_in[7];
    const float* bv    = (const float*)d_in[8];
    const float* Wo    = (const float*)d_in[9];
    const float* bo    = (const float*)d_in[10];
    float* out = (float*)d_out;

    const size_t per = (size_t)H_ * B_ * S_ * D_;   // elements per f16 tensor
    half_t* qh = (half_t*)d_ws;
    half_t* kh = qh + per;
    half_t* vh = kh + per;
    half_t* ah = vh + per;                           // 4 * 33.5 MB = 134 MB

    dim3 pblk(256), pgrd(S_ / 64, D_ / 64, H_ * B_);
    mha_proj_kernel<<<pgrd, pblk, 0, stream>>>(query, Wq, bq, qh);
    mha_proj_kernel<<<pgrd, pblk, 0, stream>>>(key_,  Wk, bk, kh);
    mha_proj_kernel<<<pgrd, pblk, 0, stream>>>(value, Wv, bv, vh);

    dim3 ablk(128), agrd(S_ / 64, H_ * B_);
    mha_attn_kernel<<<agrd, ablk, 0, stream>>>(qh, kh, vh, ah);

    dim3 oblk(256), ogrd((B_ * S_) / 64, D_ / 64);
    mha_outproj_kernel<<<ogrd, oblk, 0, stream>>>(ah, Wo, bo, out);
}
